// TripletLoss_15221364097742
// MI455X (gfx1250) — matrix operand order repaired and checked by reading, not compiled, not run on hardware
//
#include <hip/hip_runtime.h>
#include <hip/hip_bf16.h>

// Triplet loss (batch-hard) fused pipeline for MI455X (gfx1250, wave32).
// Gram matrix via v_wmma_f32_16x16x32_bf16 with bf16 hi/lo split (near-fp32),
// double-buffered GLOBAL_LOAD_ASYNC_TO_LDS staging (ASYNCcnt), fused
// distance + masked hard-mining epilogue; 4096x4096 dist matrix never hits memory.

typedef __attribute__((ext_vector_type(16))) __bf16 v16bf;
typedef __attribute__((ext_vector_type(8)))  __bf16 v8bf;
typedef __attribute__((ext_vector_type(8)))  float  v8f;

// async-to-LDS builtin takes AS(1)/AS(3) pointers to a 16B int vector
typedef int v4i __attribute__((vector_size(16)));
typedef __attribute__((address_space(1))) v4i* as1_v4i_p;
typedef __attribute__((address_space(3))) v4i* as3_v4i_p;

constexpr int NROWS = 4096;
constexpr int DIM   = 1024;
constexpr int BT    = 128;   // block tile (rows and cols)
constexpr int KC    = 32;    // K chunk per WMMA stage
constexpr int NCH   = DIM / KC;
constexpr int LSTR  = 56;    // LDS row stride in halfs (112B: 16B-aligned, conflict-free)
constexpr float MARGIN = 0.3f;
constexpr float EPSV   = 1e-12f;

#if defined(__has_builtin)
#if __has_builtin(__builtin_amdgcn_global_load_async_to_lds_b128)
#define HAS_ASYNC_LDS 1
#endif
#if __has_builtin(__builtin_amdgcn_s_wait_asynccnt)
#define HAS_WAIT_ASYNC 1
#endif
#endif

__device__ __forceinline__ void copy_b128_to_lds(const __bf16* g, __bf16* l) {
#ifdef HAS_ASYNC_LDS
    __builtin_amdgcn_global_load_async_to_lds_b128(
        (as1_v4i_p)(void*)g,
        (as3_v4i_p)(void*)l,
        0 /*imm offset*/, 0 /*cpol*/);
#else
    *(uint4*)l = *(const uint4*)g;   // sync fallback: load -> ds_store
#endif
}

__device__ __forceinline__ void wait_async_zero() {
#ifdef HAS_ASYNC_LDS
#ifdef HAS_WAIT_ASYNC
    __builtin_amdgcn_s_wait_asynccnt(0);
#else
    asm volatile("s_wait_asynccnt 0x0" ::: "memory");
#endif
#endif
}

// ---------------- prep: bf16 split planes, row sq-norms, init ap/an ----------
__global__ __launch_bounds__(256)
void prep_kernel(const float* __restrict__ X,
                 __bf16* __restrict__ hi, __bf16* __restrict__ lo,
                 float* __restrict__ sq,
                 unsigned* __restrict__ apb, unsigned* __restrict__ anb)
{
    const int row = blockIdx.x;
    const int t   = threadIdx.x;
    const float* xr = X + (size_t)row * DIM;
    float acc = 0.f;
    for (int i = t; i < DIM; i += 256) {
        float x = xr[i];
        __bf16 h = (__bf16)x;
        __bf16 l = (__bf16)(x - (float)h);
        hi[(size_t)row * DIM + i] = h;
        lo[(size_t)row * DIM + i] = l;
        acc += x * x;
    }
    __shared__ float red[256];
    red[t] = acc;
    __syncthreads();
    for (int s = 128; s > 0; s >>= 1) {
        if (t < s) red[t] += red[t + s];
        __syncthreads();
    }
    if (t == 0) {
        sq[row]  = red[0];
        apb[row] = 0u;           // max over dist>=0: 0.0f is identity
        anb[row] = 0x7F800000u;  // +inf: identity for min
    }
}

// ---------------- fused Gram + distance + hard mining ------------------------
__global__ __launch_bounds__(256)
void gram_kernel(const __bf16* __restrict__ hi, const __bf16* __restrict__ lo,
                 const float* __restrict__ sq, const int* __restrict__ tgt,
                 unsigned* __restrict__ apb, unsigned* __restrict__ anb)
{
    const int rb = blockIdx.y, cb = blockIdx.x;
    if (cb < rb) return;   // symmetric: upper triangle only (uniform per block)

    // [buffer][plane: AH, AL, BH, BL]
    __shared__ __bf16 stage[2][4][BT * LSTR];
    __shared__ float sqR[BT], sqC[BT];
    __shared__ int   tgR[BT], tgC[BT];

    const int tid  = threadIdx.x;
    const int lane = tid & 31;
    const int wave = tid >> 5;    // 8 waves
    const int wm   = wave >> 2;   // 0..1 -> 64-row strip
    const int wn   = wave & 3;    // 0..3 -> 32-col strip

    if (tid < BT) {
        sqR[tid] = sq[rb * BT + tid];  tgR[tid] = tgt[rb * BT + tid];
        sqC[tid] = sq[cb * BT + tid];  tgC[tid] = tgt[cb * BT + tid];
    }

    v8f c[4][2];
    {
        v8f z;
        #pragma unroll
        for (int i = 0; i < 8; i++) z[i] = 0.f;
        #pragma unroll
        for (int mi = 0; mi < 4; mi++)
            #pragma unroll
            for (int ni = 0; ni < 2; ni++) c[mi][ni] = z;
    }

    // staging geometry: 256 threads, each moves 16 halfs (32B) per plane
    const int ldRow  = tid >> 1;
    const int ldHalf = (tid & 1) * 16;
    const size_t gA  = (size_t)(rb * BT + ldRow) * DIM + ldHalf;
    const size_t gB  = (size_t)(cb * BT + ldRow) * DIM + ldHalf;
    const int    lofs = ldRow * LSTR + ldHalf;

    // fragment geometry (wave32 WMMA layouts)
    const int halfsel = lane >> 4;      // 0 / 1
    const int lm      = lane & 15;
    const int aC0     = halfsel * 8;    // A: K in {aC0..aC0+7} U {aC0+16..aC0+23}
    const int bC0     = halfsel * 8;    // B: K in {bC0..bC0+7} U {bC0+16..bC0+23}

    const __bf16* gsrc[4] = { hi + gA, lo + gA, hi + gB, lo + gB };

    // prologue: fill buffer 0 with chunk 0
    #pragma unroll
    for (int p = 0; p < 4; p++) {
        __bf16* l = &stage[0][p][lofs];
        copy_b128_to_lds(gsrc[p],     l);
        copy_b128_to_lds(gsrc[p] + 8, l + 8);
    }

    for (int it = 0; it < NCH; ++it) {
        const int buf = it & 1;
        wait_async_zero();     // this wave's async loads into `buf` landed in LDS
        __syncthreads();       // all waves landed; all reads of `buf^1` retired

        if (it + 1 < NCH) {    // stream next chunk into the other buffer
            const int kn = (it + 1) * KC;
            #pragma unroll
            for (int p = 0; p < 4; p++) {
                __bf16* l = &stage[buf ^ 1][p][lofs];
                copy_b128_to_lds(gsrc[p] + kn,     l);
                copy_b128_to_lds(gsrc[p] + kn + 8, l + 8);
            }
        }

        const __bf16* AsH = stage[buf][0];
        const __bf16* AsL = stage[buf][1];
        const __bf16* BsH = stage[buf][2];
        const __bf16* BsL = stage[buf][3];

        v16bf aH[4], aL[4], bH[2], bL[2];
        #pragma unroll
        for (int mi = 0; mi < 4; mi++) {
            const int r = wm * 64 + mi * 16 + lm;
            const __bf16* p = AsH + r * LSTR + aC0;
            v8bf x0 = *(const v8bf*)p;
            v8bf x1 = *(const v8bf*)(p + 16);
            aH[mi] = __builtin_shufflevector(x0, x1, 0,1,2,3,4,5,6,7,8,9,10,11,12,13,14,15);
            const __bf16* q = AsL + r * LSTR + aC0;
            v8bf y0 = *(const v8bf*)q;
            v8bf y1 = *(const v8bf*)(q + 16);
            aL[mi] = __builtin_shufflevector(y0, y1, 0,1,2,3,4,5,6,7,8,9,10,11,12,13,14,15);
        }
        #pragma unroll
        for (int ni = 0; ni < 2; ni++) {
            const int r = wn * 32 + ni * 16 + lm;
            const __bf16* p = BsH + r * LSTR + bC0;
            v8bf x0 = *(const v8bf*)p;
            v8bf x1 = *(const v8bf*)(p + 16);
            bH[ni] = __builtin_shufflevector(x0, x1, 0,1,2,3,4,5,6,7,8,9,10,11,12,13,14,15);
            const __bf16* q = BsL + r * LSTR + bC0;
            v8bf y0 = *(const v8bf*)q;
            v8bf y1 = *(const v8bf*)(q + 16);
            bL[ni] = __builtin_shufflevector(y0, y1, 0,1,2,3,4,5,6,7,8,9,10,11,12,13,14,15);
        }

        #pragma unroll
        for (int mi = 0; mi < 4; mi++)
            #pragma unroll
            for (int ni = 0; ni < 2; ni++) {
                c[mi][ni] = __builtin_amdgcn_wmma_f32_16x16x32_bf16(
                    false, aH[mi], false, bH[ni], (short)0, c[mi][ni], false, false);
                c[mi][ni] = __builtin_amdgcn_wmma_f32_16x16x32_bf16(
                    false, aH[mi], false, bL[ni], (short)0, c[mi][ni], false, false);
                c[mi][ni] = __builtin_amdgcn_wmma_f32_16x16x32_bf16(
                    false, aL[mi], false, bH[ni], (short)0, c[mi][ni], false, false);
            }
    }

    // -------- epilogue: dist = sqrt(clip(sq_m + sq_n - 2 dot)), hard mining --
    float colAp[2] = {0.f, 0.f};
    float colAn[2] = {__builtin_huge_valf(), __builtin_huge_valf()};

    #pragma unroll
    for (int mi = 0; mi < 4; mi++) {
        float rAp[8], rAn[8];
        #pragma unroll
        for (int r = 0; r < 8; r++) { rAp[r] = 0.f; rAn[r] = __builtin_huge_valf(); }

        #pragma unroll
        for (int ni = 0; ni < 2; ni++) {
            const int nloc = wn * 32 + ni * 16 + lm;
            const float sn = sqC[nloc];
            const int   tn = tgC[nloc];
            #pragma unroll
            for (int r = 0; r < 8; r++) {
                const int mloc = wm * 64 + mi * 16 + halfsel * 8 + r;
                float d2 = sqR[mloc] + sn - 2.0f * c[mi][ni][r];
                float d  = sqrtf(fmaxf(d2, EPSV));
                const bool same = (tgR[mloc] == tn);
                const float ap = same ? d : 0.0f;
                const float an = same ? __builtin_huge_valf() : d;
                rAp[r] = fmaxf(rAp[r], ap);
                rAn[r] = fminf(rAn[r], an);
                colAp[ni] = fmaxf(colAp[ni], ap);
                colAn[ni] = fminf(colAn[ni], an);
            }
        }
        // reduce across the 16 lanes sharing each row (xor masks <16 stay in half)
        #pragma unroll
        for (int r = 0; r < 8; r++) {
            float ap = rAp[r], an = rAn[r];
            for (int s = 1; s < 16; s <<= 1) {
                ap = fmaxf(ap, __shfl_xor(ap, s, 32));
                an = fminf(an, __shfl_xor(an, s, 32));
            }
            if (lm == 0) {
                const int grow = rb * BT + wm * 64 + mi * 16 + halfsel * 8 + r;
                atomicMax(apb + grow, __float_as_uint(ap));
                atomicMin(anb + grow, __float_as_uint(an));
            }
        }
    }
    // column-direction (transpose) contribution: rows = global columns
    #pragma unroll
    for (int ni = 0; ni < 2; ni++) {
        float ap = colAp[ni], an = colAn[ni];
        ap = fmaxf(ap, __shfl_xor(ap, 16, 32));
        an = fminf(an, __shfl_xor(an, 16, 32));
        if (halfsel == 0) {
            const int gcol = cb * BT + wn * 32 + ni * 16 + lm;
            atomicMax(apb + gcol, __float_as_uint(ap));
            atomicMin(anb + gcol, __float_as_uint(an));
        }
    }
}

// ---------------- finalize: loss + precision ---------------------------------
__global__ __launch_bounds__(1024)
void finalize_kernel(const unsigned* __restrict__ apb,
                     const unsigned* __restrict__ anb,
                     float* __restrict__ out)
{
    const int t = threadIdx.x;
    float lsum = 0.f, csum = 0.f;
    for (int i = t; i < NROWS; i += 1024) {
        const float ap = __uint_as_float(apb[i]);
        const float an = __uint_as_float(anb[i]);
        lsum += fmaxf(MARGIN - (an - ap), 0.f);
        csum += (an > ap) ? 1.f : 0.f;
    }
    __shared__ float r1[1024], r2[1024];
    r1[t] = lsum; r2[t] = csum;
    __syncthreads();
    for (int s = 512; s > 0; s >>= 1) {
        if (t < s) { r1[t] += r1[t + s]; r2[t] += r2[t + s]; }
        __syncthreads();
    }
    if (t == 0) {
        out[0] = r1[0] / (float)NROWS;
        out[1] = r2[0] / (float)NROWS;
    }
}

// ---------------- launcher ---------------------------------------------------
extern "C" void kernel_launch(void* const* d_in, const int* in_sizes, int n_in,
                              void* d_out, int out_size, void* d_ws, size_t ws_size,
                              hipStream_t stream) {
    (void)in_sizes; (void)n_in; (void)out_size; (void)ws_size;
    const float* X   = (const float*)d_in[0];
    const int*   tgt = (const int*)d_in[1];
    float*       out = (float*)d_out;

    char* w = (char*)d_ws;
    const size_t planeBytes = (size_t)NROWS * DIM * sizeof(__bf16); // 8 MB
    __bf16*   hi  = (__bf16*)(w);
    __bf16*   lo  = (__bf16*)(w + planeBytes);
    float*    sq  = (float*)(w + 2 * planeBytes);
    unsigned* apb = (unsigned*)(w + 2 * planeBytes + NROWS * sizeof(float));
    unsigned* anb = apb + NROWS;

    prep_kernel<<<NROWS, 256, 0, stream>>>(X, hi, lo, sq, apb, anb);

    dim3 grid(NROWS / BT, NROWS / BT);  // 32 x 32, lower triangle early-outs
    gram_kernel<<<grid, 256, 0, stream>>>(hi, lo, sq, tgt, apb, anb);

    finalize_kernel<<<1, 1024, 0, stream>>>(apb, anb, out);
}
